// CGFN_block_90348932038925
// MI455X (gfx1250) — compile-verified, matrix operand order checked by reading, not run on hardware
//
#include <hip/hip_runtime.h>
#include <hip/hip_bf16.h>

// ---------------------------------------------------------------------------
// DyGCN block for MI455X (gfx1250, wave32, WMMA).
// B=16, N=512, C=64, T=12, H=64.
// Heavy GEMMs (scores = xt@xt^T, adj@xt, enh@xt, 64x64 projections) run as a
// flash-style fused kernel using v_wmma_f32_16x16x32_bf16 (f32 accumulate),
// with async global->LDS tile staging (ASYNCcnt) and permlane16 reductions.
// ---------------------------------------------------------------------------

#define DEVI __device__ __forceinline__

typedef __attribute__((ext_vector_type(16))) __bf16 v16bf;
typedef __attribute__((ext_vector_type(8)))  float  v8f;

static constexpr int B_ = 16, N_ = 512, C_ = 64, T_ = 12, H_ = 64;

// K-offset map for 16-bit A-fragment (16x32, wave32):
// lanes 0-15 hold K in {0..7, 16..23}; lanes 16-31 hold K in {8..15, 24..31}.
DEVI int kmap(int e, int half) {
  return (e < 8) ? (half * 8 + e) : (16 + half * 8 + (e - 8));
}

DEVI v8f wmma_bf(v16bf a, v16bf b, v8f c) {
  return __builtin_amdgcn_wmma_f32_16x16x32_bf16(false, a, false, b,
                                                 (short)0, c, false, false);
}

// xor-butterfly within 16-lane rows on the VALU (co-executes with XDL WMMA).
constexpr unsigned selLo(int m) {
  unsigned s = 0;
  for (int i = 0; i < 8; ++i) s |= ((unsigned)((i ^ m) & 0xF)) << (4 * i);
  return s;
}
constexpr unsigned selHi(int m) {
  unsigned s = 0;
  for (int i = 8; i < 16; ++i) s |= ((unsigned)((i ^ m) & 0xF)) << (4 * (i - 8));
  return s;
}

template <int M>
DEVI float xor16_max(float v) {
#if __has_builtin(__builtin_amdgcn_permlane16)
  int r = __builtin_amdgcn_permlane16(__float_as_int(v), __float_as_int(v),
                                      (int)selLo(M), (int)selHi(M), false, false);
  return fmaxf(v, __int_as_float(r));
#else
  return fmaxf(v, __shfl_xor(v, M, 32));
#endif
}

// ---------------------------------------------------------------------------
// Prep: enh[t,n,m] = static_adj[n,m] * sigmoid(masks[t,n,m]) in bf16
// ---------------------------------------------------------------------------
__global__ void k_enh(const float* __restrict__ A, const float* __restrict__ masks,
                      __bf16* __restrict__ enh, int total) {
  int i = blockIdx.x * 256 + threadIdx.x;
  if (i >= total) return;
  int nm = i % (N_ * N_);
  float s = 1.f / (1.f + __expf(-masks[i]));
  enh[i] = (__bf16)(A[nm] * s);
}

// ---------------------------------------------------------------------------
// Temporal attention (tiny: T=12)
// ---------------------------------------------------------------------------
__global__ void k_tmp1(const float* __restrict__ x, const float* __restrict__ U1,
                       float* __restrict__ tmp1) {
  int bt = blockIdx.x;          // b*T + t
  int b = bt / T_, t = bt % T_;
  int c = threadIdx.x;          // 64
  float acc = 0.f;
  for (int n = 0; n < N_; ++n)
    acc += x[(((size_t)b * N_ + n) * C_ + c) * T_ + t] * U1[n];
  tmp1[bt * C_ + c] = acc;
}

__global__ void k_rhs(const float* __restrict__ x, const float* __restrict__ U3,
                      float* __restrict__ rhs) {
  int i = blockIdx.x * 256 + threadIdx.x;   // over B*N*T
  if (i >= B_ * N_ * T_) return;
  int t = i % T_;
  int n = (i / T_) % N_;
  int b = i / (T_ * N_);
  float acc = 0.f;
  for (int c = 0; c < C_; ++c)
    acc += U3[c] * x[(((size_t)b * N_ + n) * C_ + c) * T_ + t];
  rhs[i] = acc;                             // [b][n][t]
}

__global__ void k_lhs(const float* __restrict__ tmp1, const float* __restrict__ U2,
                      float* __restrict__ lhs) {
  __shared__ float st[C_];
  int bt = blockIdx.x;
  int n = threadIdx.x;                      // 512
  if (n < C_) st[n] = tmp1[bt * C_ + n];
  __syncthreads();
  float acc = 0.f;
#pragma unroll
  for (int c = 0; c < C_; ++c) acc += st[c] * U2[c * N_ + n];
  lhs[bt * N_ + n] = acc;                   // [b][t][n]
}

__global__ void k_att(const float* __restrict__ lhs, const float* __restrict__ rhs,
                      const float* __restrict__ be, const float* __restrict__ Ve,
                      float* __restrict__ At) {
  int b = blockIdx.x;
  int tid = threadIdx.x;
  __shared__ float sp[T_ * T_], sE[T_ * T_];
  if (tid < T_ * T_) {
    int j = tid / T_, k = tid % T_;
    float acc = 0.f;
    for (int n = 0; n < N_; ++n)
      acc += lhs[(b * T_ + j) * N_ + n] * rhs[(b * N_ + n) * T_ + k];
    float v = acc + be[j * T_ + k];
    sp[tid] = 1.f / (1.f + __expf(-v));
  }
  __syncthreads();
  if (tid < T_ * T_) {
    int i = tid / T_, k = tid % T_;
    float acc = 0.f;
#pragma unroll
    for (int j = 0; j < T_; ++j) acc += Ve[i * T_ + j] * sp[j * T_ + k];
    sE[tid] = acc;
  }
  __syncthreads();
  if (tid < T_) {                            // softmax over axis i
    int k = tid;
    float mx = -3.0e38f;
    for (int i = 0; i < T_; ++i) mx = fmaxf(mx, sE[i * T_ + k]);
    float s = 0.f, ev[T_];
    for (int i = 0; i < T_; ++i) { ev[i] = __expf(sE[i * T_ + k] - mx); s += ev[i]; }
    for (int i = 0; i < T_; ++i) At[(b * T_ + i) * T_ + k] = ev[i] / s;
  }
}

// xt_bf[b,s,n,c] = bf16( sum_t x[b,n,c,t] * At[b,t,s] )
__global__ void k_xt(const float* __restrict__ x, const float* __restrict__ At,
                     __bf16* __restrict__ xtbf) {
  __shared__ float sA[T_ * T_];
  int b = blockIdx.x;
  int tid = threadIdx.x;
  if (tid < T_ * T_) sA[tid] = At[b * T_ * T_ + tid];
  __syncthreads();
  int m = blockIdx.y * 256 + tid;            // over N*C
  int n = m >> 6, c = m & 63;
  const float* xp = x + (((size_t)b * N_ + n) * C_ + c) * T_;
  float xv[T_];
#pragma unroll
  for (int t = 0; t < T_; ++t) xv[t] = xp[t];
#pragma unroll
  for (int s = 0; s < T_; ++s) {
    float acc = 0.f;
#pragma unroll
    for (int t = 0; t < T_; ++t) acc += xv[t] * sA[t * T_ + s];
    xtbf[(((size_t)b * T_ + s) * N_ + n) * C_ + c] = (__bf16)acc;
  }
}

// ---------------------------------------------------------------------------
// Fused GCN: per (b,t): scores -> online softmax -> adj@xt (dynamic),
// enh@xt (static), three 64x64 projections, fused combine. All via bf16 WMMA.
// Grid: (N/64, T, B); block 128 = 4 waves; each wave owns 16 rows.
// ---------------------------------------------------------------------------
__global__ __launch_bounds__(128)
void k_fused_gcn(const __bf16* __restrict__ xt,    // [B,T,N,C] bf16
                 const __bf16* __restrict__ enh,   // [T,N,N]
                 const float* __restrict__ W0,     // [H,C]
                 const float* __restrict__ Ws,     // [T,H,C]
                 const float* __restrict__ Wd,     // [T,H,C]
                 const float* __restrict__ b0,
                 const float* __restrict__ bs,     // [T,H]
                 const float* __restrict__ bd,     // [T,H]
                 float* __restrict__ X)            // [B,H,N,T]
{
  const int b = blockIdx.z, t = blockIdx.y;
  const int tid = threadIdx.x;
  const int wave = tid >> 5;
  const int lane = tid & 31;
  const int half = lane >> 4;
  const int l16 = lane & 15;
  const int row0 = blockIdx.x * 64 + wave * 16;

  __shared__ __bf16 sW0[64 * 64];
  __shared__ __bf16 sWs[64 * 64];
  __shared__ __bf16 sWd[64 * 64];
  __shared__ __align__(16) __bf16 sXt[32 * 64];  // 32-row tile (async dest)
  __shared__ __bf16 sB[4][16 * 64];              // per-wave layout bounce

  for (int i = tid; i < 64 * 64; i += 128) {
    sW0[i] = (__bf16)W0[i];
    sWs[i] = (__bf16)Ws[t * 64 * 64 + i];
    sWd[i] = (__bf16)Wd[t * 64 * 64 + i];
  }

  const __bf16* xtb = xt + ((size_t)(b * T_ + t)) * N_ * C_;
  const __bf16* enhT = enh + (size_t)t * N_ * N_;

  // A fragments for this wave's 16 rows (K = channel dim, 2 blocks of 32)
  v16bf aRow[2];
#pragma unroll
  for (int ks = 0; ks < 2; ++ks)
#pragma unroll
    for (int e = 0; e < 16; ++e)
      aRow[ks][e] = xtb[(size_t)(row0 + l16) * C_ + ks * 32 + kmap(e, half)];

  // constant ones B-fragment: row-sum of P via the WMMA unit itself
  v16bf bOne;
#pragma unroll
  for (int e = 0; e < 16; ++e) bOne[e] = (__bf16)1.0f;

  const v8f vzero = {0.f, 0.f, 0.f, 0.f, 0.f, 0.f, 0.f, 0.f};
  v8f dAcc[5], sAcc[4];                      // dAcc[4] = running softmax denom
#pragma unroll
  for (int nt = 0; nt < 5; ++nt) dAcc[nt] = vzero;
#pragma unroll
  for (int nt = 0; nt < 4; ++nt) sAcc[nt] = vzero;
  float rowmax[8];
#pragma unroll
  for (int r = 0; r < 8; ++r) rowmax[r] = -3.0e38f;

  // per-thread async-copy addresses: contiguous 4KB tile, 32B per thread
  const unsigned ldsOff =
      (unsigned)(uintptr_t)(&sXt[0]) + (unsigned)tid * 32u;

  for (int jj = 0; jj < 16; ++jj) {        // 32-column blocks over m = 512
    __syncthreads();
    {
      // CDNA5 async global->LDS copy (tracked by ASYNCcnt)
      unsigned long long ga =
          (unsigned long long)(uintptr_t)(xtb + (size_t)jj * 32 * C_) +
          (unsigned long long)tid * 32ull;
      asm volatile(
          "global_load_async_to_lds_b128 %0, %1, off\n\t"
          "global_load_async_to_lds_b128 %0, %1, off offset:16\n\t"
          "s_wait_asynccnt 0x0"
          :: "v"(ldsOff), "v"(ga) : "memory");
    }
    if (jj + 1 < 16)
      __builtin_prefetch(xtb + (size_t)(jj + 1) * 32 * C_, 0, 1);
    __syncthreads();

    // --- scores: S[16 rows x 32 cols] = A_rows @ Xt_tile^T -----------------
    v8f s0 = vzero, s1 = vzero;
#pragma unroll
    for (int ks = 0; ks < 2; ++ks) {
      v16bf b0f, b1f;
#pragma unroll
      for (int e = 0; e < 16; ++e) {
        int c = ks * 32 + kmap(e, half);
        b0f[e] = sXt[(0 * 16 + l16) * 64 + c];
        b1f[e] = sXt[(1 * 16 + l16) * 64 + c];
      }
      s0 = wmma_bf(aRow[ks], b0f, s0);
      s1 = wmma_bf(aRow[ks], b1f, s1);
    }

    // static adjacency A-fragment for this k-block (global bf16 gather)
    v16bf aE;
#pragma unroll
    for (int e = 0; e < 16; ++e)
      aE[e] = enhT[(size_t)(row0 + l16) * N_ + jj * 32 + kmap(e, half)];

    // --- online softmax (scale 1/sqrt(C) = 0.125); sum comes from WMMA -----
#pragma unroll
    for (int r = 0; r < 8; ++r) {
      float v0 = s0[r] * 0.125f, v1 = s1[r] * 0.125f;
      float tmax = fmaxf(v0, v1);
      tmax = xor16_max<1>(tmax);
      tmax = xor16_max<2>(tmax);
      tmax = xor16_max<4>(tmax);
      tmax = xor16_max<8>(tmax);
      float mnew = fmaxf(rowmax[r], tmax);
      float scale = __expf(rowmax[r] - mnew);
      rowmax[r] = mnew;
      float p0 = __expf(v0 - mnew), p1 = __expf(v1 - mnew);
#pragma unroll
      for (int nt = 0; nt < 5; ++nt) dAcc[nt][r] *= scale;   // incl. denom
      int mrow = r + half * 8;             // D-layout row
      sB[wave][mrow * 32 + l16] = (__bf16)p0;
      sB[wave][mrow * 32 + 16 + l16] = (__bf16)p1;
    }
    __syncthreads();

    // P in A-fragment layout (K = 32)
    v16bf aP;
#pragma unroll
    for (int e = 0; e < 16; ++e)
      aP[e] = sB[wave][l16 * 32 + kmap(e, half)];

    // dAcc += P @ Xt_tile ; sAcc += Enh_tile @ Xt_tile (shared B fragments);
    // dAcc[4] += P @ ones  (softmax denominator)
#pragma unroll
    for (int nt = 0; nt < 4; ++nt) {
      v16bf bC;
#pragma unroll
      for (int e = 0; e < 16; ++e)
        bC[e] = sXt[kmap(e, half) * 64 + nt * 16 + l16];
      dAcc[nt] = wmma_bf(aP, bC, dAcc[nt]);
      sAcc[nt] = wmma_bf(aE, bC, sAcc[nt]);
    }
    dAcc[4] = wmma_bf(aP, bOne, dAcc[4]);
  }

  float invs[8];
#pragma unroll
  for (int r = 0; r < 8; ++r) invs[r] = 1.f / dAcc[4][r];

  v8f outAcc[4];
#pragma unroll
  for (int nt = 0; nt < 4; ++nt) outAcc[nt] = vzero;

  // ---- projection helper: outAcc += A(16x64, bounced) @ W^T ---------------
  auto proj = [&](const __bf16* A, const __bf16* W) {
#pragma unroll
    for (int ks = 0; ks < 2; ++ks) {
      v16bf a;
#pragma unroll
      for (int e = 0; e < 16; ++e)
        a[e] = A[l16 * 64 + ks * 32 + kmap(e, half)];
#pragma unroll
      for (int nt = 0; nt < 4; ++nt) {
        v16bf bw;
#pragma unroll
        for (int e = 0; e < 16; ++e)
          bw[e] = W[(nt * 16 + l16) * 64 + ks * 32 + kmap(e, half)];
        outAcc[nt] = wmma_bf(a, bw, outAcc[nt]);
      }
    }
  };

  // dynamic branch: (0.475 / rowsum) * dAcc @ Wd^T
  __syncthreads();
#pragma unroll
  for (int nt = 0; nt < 4; ++nt)
#pragma unroll
    for (int r = 0; r < 8; ++r)
      sB[wave][(r + half * 8) * 64 + nt * 16 + l16] =
          (__bf16)(dAcc[nt][r] * invs[r] * 0.475f);
  __syncthreads();
  proj(sB[wave], sWd);

  // static branch: 0.475 * sAcc @ Ws^T
  __syncthreads();
#pragma unroll
  for (int nt = 0; nt < 4; ++nt)
#pragma unroll
    for (int r = 0; r < 8; ++r)
      sB[wave][(r + half * 8) * 64 + nt * 16 + l16] = (__bf16)(sAcc[nt][r] * 0.475f);
  __syncthreads();
  proj(sB[wave], sWs);

  // shared linear: 0.05 * xt_rows @ W0^T (rows already held as A-fragments)
  __syncthreads();
#pragma unroll
  for (int ks = 0; ks < 2; ++ks)
#pragma unroll
    for (int e = 0; e < 16; ++e)
      sB[wave][l16 * 64 + ks * 32 + kmap(e, half)] =
          (__bf16)(0.05f * (float)aRow[ks][e]);
  __syncthreads();
  proj(sB[wave], sW0);

  // combine biases, write directly in X = [B,H,N,T] layout for conv stage
#pragma unroll
  for (int nt = 0; nt < 4; ++nt) {
    int h = nt * 16 + l16;
    float bias = 0.05f * b0[h] + 0.475f * (bs[t * H_ + h] + bd[t * H_ + h]);
#pragma unroll
    for (int r = 0; r < 8; ++r) {
      int n = row0 + r + half * 8;
      X[(((size_t)b * H_ + h) * N_ + n) * T_ + t] = outAcc[nt][r] + bias;
    }
  }
}

// ---------------------------------------------------------------------------
// Multi-scale gated temporal convs
// ---------------------------------------------------------------------------
__global__ void k_conv(const float* __restrict__ X, const float* __restrict__ w,
                       const float* __restrict__ cb, float* __restrict__ xc,
                       int K, int Tk) {
  int n = blockIdx.x * 128 + threadIdx.x;
  int o = blockIdx.y;                      // 0..127 (2H)
  int b = blockIdx.z;
  int g = o >> 5;                          // feature group (groups=4)
  float acc[10];
  for (int t = 0; t < Tk; ++t) acc[t] = cb[o];
  for (int i = 0; i < 16; ++i) {
    const float* xp = X + (((size_t)b * H_ + g * 16 + i) * N_ + n) * T_;
    float xv[T_];
#pragma unroll
    for (int t = 0; t < T_; ++t) xv[t] = xp[t];
    for (int kk = 0; kk < K; ++kk) {
      float wv = w[(o * 16 + i) * K + kk];
      for (int t = 0; t < Tk; ++t) acc[t] += xv[t + kk] * wv;
    }
  }
  float* op = xc + (((size_t)b * 128 + o) * N_ + n) * Tk;
  for (int t = 0; t < Tk; ++t) op[t] = acc[t];
}

__global__ void k_gnstats(const float* __restrict__ xc, float* __restrict__ stats,
                          int Tk) {
  int gy = blockIdx.x;                     // 0..7: half*4 + group
  int b = blockIdx.y;
  int tid = threadIdx.x;
  int chbase = (gy >> 2) * 64 + (gy & 3) * 16;
  int count = 16 * N_ * Tk;
  float s1 = 0.f, s2 = 0.f;
  for (int li = tid; li < count; li += 256) {
    int ci = li / (N_ * Tk);
    int rem = li % (N_ * Tk);
    float v = xc[(size_t)(b * 128 + chbase + ci) * N_ * Tk + rem];
    s1 += v; s2 += v * v;
  }
  __shared__ float r1[256], r2[256];
  r1[tid] = s1; r2[tid] = s2;
  __syncthreads();
  for (int s = 128; s > 0; s >>= 1) {
    if (tid < s) { r1[tid] += r1[tid + s]; r2[tid] += r2[tid + s]; }
    __syncthreads();
  }
  if (tid == 0) {
    float mu = r1[0] / count;
    float var = r2[0] / count - mu * mu;
    stats[(b * 8 + gy) * 2 + 0] = mu;
    stats[(b * 8 + gy) * 2 + 1] = rsqrtf(var + 1e-5f);
  }
}

__global__ void k_gnact(const float* __restrict__ xc, const float* __restrict__ stats,
                        const float* __restrict__ pw, const float* __restrict__ pb,
                        const float* __restrict__ qw, const float* __restrict__ qb,
                        float* __restrict__ concat, int Tk, int off, int total) {
  int e = blockIdx.x * 256 + threadIdx.x;
  if (e >= total) return;
  int tau = e % Tk;
  int n = (e / Tk) % N_;
  int h = (e / (Tk * N_)) % H_;
  int b = e / (Tk * N_ * H_);
  int gp = h >> 4;
  float xp = xc[(((size_t)b * 128 + h) * N_ + n) * Tk + tau];
  float xq = xc[(((size_t)b * 128 + 64 + h) * N_ + n) * Tk + tau];
  float mp = stats[(b * 8 + gp) * 2], rp = stats[(b * 8 + gp) * 2 + 1];
  float mq = stats[(b * 8 + 4 + gp) * 2], rq = stats[(b * 8 + 4 + gp) * 2 + 1];
  float p = (xp - mp) * rp * pw[h] + pb[h];
  float q = (xq - mq) * rq * qw[h] + qb[h];
  float sp_ = (p > 20.f) ? p : log1pf(__expf(p));
  float mish = p * tanhf(sp_);
  float gate = fminf(fmaxf(q, 0.f), 1.f);
  concat[(((size_t)b * H_ + h) * N_ + n) * 28 + off + tau] = mish * gate;
}

__global__ void k_fc(const float* __restrict__ concat, const float* __restrict__ fcw,
                     const float* __restrict__ fcb, float* __restrict__ tco) {
  int i = blockIdx.x * 256 + threadIdx.x;  // over B*H*N
  if (i >= B_ * H_ * N_) return;
  const float* row = concat + (size_t)i * 28;
  float rv[28];
#pragma unroll
  for (int k = 0; k < 28; ++k) rv[k] = row[k];
  float* op = tco + (size_t)i * T_;
#pragma unroll
  for (int o = 0; o < T_; ++o) {
    float acc = fcb[o];
#pragma unroll
    for (int k = 0; k < 28; ++k) acc += rv[k] * fcw[o * 28 + k];
    op[o] = fmaxf(acc, 0.f);
  }
}

// ---------------------------------------------------------------------------
// Residual projection + final ReLU + LayerNorm over H; output [B,N,H,T]
// ---------------------------------------------------------------------------
__global__ __launch_bounds__(128)
void k_resln(const float* __restrict__ x, const float* __restrict__ tco,
             const float* __restrict__ rw, const float* __restrict__ rb,
             const float* __restrict__ lw, const float* __restrict__ lb,
             float* __restrict__ out) {
  int n = blockIdx.x, b = blockIdx.y, tid = threadIdx.x;
  __shared__ float sx[C_ * T_];
  __shared__ float sy[H_ * T_];
  __shared__ float smu[T_], srs[T_];
  const float* xp = x + ((size_t)b * N_ + n) * C_ * T_;
  for (int i = tid; i < C_ * T_; i += 128) sx[i] = xp[i];
  __syncthreads();
  for (int i = tid; i < H_ * T_; i += 128) {
    int h = i / T_, t = i % T_;
    float acc = rb[h];
#pragma unroll
    for (int c = 0; c < C_; ++c) acc += sx[c * T_ + t] * rw[h * C_ + c];
    float y = acc + tco[(((size_t)b * H_ + h) * N_ + n) * T_ + t];
    sy[i] = fmaxf(y, 0.f);
  }
  __syncthreads();
  if (tid < T_) {
    int t = tid;
    float s1 = 0.f, s2 = 0.f;
    for (int h = 0; h < H_; ++h) { float v = sy[h * T_ + t]; s1 += v; s2 += v * v; }
    float mu = s1 / (float)H_;
    float var = s2 / (float)H_ - mu * mu;
    smu[t] = mu; srs[t] = rsqrtf(var + 1e-5f);
  }
  __syncthreads();
  float* op = out + ((size_t)b * N_ + n) * H_ * T_;
  for (int i = tid; i < H_ * T_; i += 128) {
    int h = i / T_, t = i % T_;
    op[h * T_ + t] = (sy[i] - smu[t]) * srs[t] * lw[h] + lb[h];
  }
}

// ---------------------------------------------------------------------------
extern "C" void kernel_launch(void* const* d_in, const int* in_sizes, int n_in,
                              void* d_out, int out_size, void* d_ws, size_t ws_size,
                              hipStream_t stream) {
  const float* x    = (const float*)d_in[0];
  const float* adj  = (const float*)d_in[1];
  const float* U1   = (const float*)d_in[2];
  const float* U2   = (const float*)d_in[3];
  const float* U3   = (const float*)d_in[4];
  const float* be   = (const float*)d_in[5];
  const float* Ve   = (const float*)d_in[6];
  const float* W0   = (const float*)d_in[7];
  const float* b0   = (const float*)d_in[8];
  const float* masks= (const float*)d_in[9];
  const float* Ws   = (const float*)d_in[10];
  const float* bs   = (const float*)d_in[11];
  const float* Wd   = (const float*)d_in[12];
  const float* bd   = (const float*)d_in[13];
  const float* convw[4] = {(const float*)d_in[14], (const float*)d_in[15],
                           (const float*)d_in[16], (const float*)d_in[17]};
  const float* conv_b = (const float*)d_in[18];
  const float* gn_pw  = (const float*)d_in[19];
  const float* gn_pb  = (const float*)d_in[20];
  const float* gn_qw  = (const float*)d_in[21];
  const float* gn_qb  = (const float*)d_in[22];
  const float* fc_w   = (const float*)d_in[23];
  const float* fc_b   = (const float*)d_in[24];
  const float* res_w  = (const float*)d_in[25];
  const float* res_b  = (const float*)d_in[26];
  const float* ln_w   = (const float*)d_in[27];
  const float* ln_b   = (const float*)d_in[28];
  (void)in_sizes; (void)n_in; (void)out_size; (void)ws_size;

  char* wsb = (char*)d_ws;
  size_t cur = 0;
  auto alloc = [&](size_t bytes) {
    size_t o = cur;
    cur = (cur + bytes + 255) & ~(size_t)255;
    return wsb + o;
  };
  float*  At     = (float*)alloc((size_t)B_ * T_ * T_ * 4);
  float*  tmp1   = (float*)alloc((size_t)B_ * T_ * C_ * 4);
  float*  lhs    = (float*)alloc((size_t)B_ * T_ * N_ * 4);
  float*  rhs    = (float*)alloc((size_t)B_ * N_ * T_ * 4);
  __bf16* xtbf   = (__bf16*)alloc((size_t)B_ * T_ * N_ * C_ * 2);
  float*  X      = (float*)alloc((size_t)B_ * H_ * N_ * T_ * 4);
  float*  tco    = (float*)alloc((size_t)B_ * H_ * N_ * T_ * 4);
  float*  concat = (float*)alloc((size_t)B_ * H_ * N_ * 28 * 4);
  float*  xc     = (float*)alloc((size_t)B_ * 128 * N_ * 10 * 4);
  float*  stats  = (float*)alloc((size_t)B_ * 8 * 2 * 4);
  __bf16* enh    = (__bf16*)alloc((size_t)T_ * N_ * N_ * 2);

  // static-branch enhanced adjacency (bf16)
  int enhTot = T_ * N_ * N_;
  k_enh<<<(enhTot + 255) / 256, 256, 0, stream>>>(adj, masks, enh, enhTot);

  // temporal attention
  k_tmp1<<<B_ * T_, C_, 0, stream>>>(x, U1, tmp1);
  k_rhs<<<(B_ * N_ * T_ + 255) / 256, 256, 0, stream>>>(x, U3, rhs);
  k_lhs<<<B_ * T_, N_, 0, stream>>>(tmp1, U2, lhs);
  k_att<<<B_, 256, 0, stream>>>(lhs, rhs, be, Ve, At);
  k_xt<<<dim3(B_, (N_ * C_) / 256), 256, 0, stream>>>(x, At, xtbf);

  // fused GCN (WMMA heavy path)
  k_fused_gcn<<<dim3(N_ / 64, T_, B_), 128, 0, stream>>>(
      xtbf, enh, W0, Ws, Wd, b0, bs, bd, X);

  // multi-scale gated temporal convs
  const int Ks[4] = {3, 5, 7, 9};
  const int Tks[4] = {10, 8, 6, 4};
  const int offs[4] = {0, 10, 18, 24};
  for (int s = 0; s < 4; ++s) {
    k_conv<<<dim3(N_ / 128, 128, B_), 128, 0, stream>>>(
        X, convw[s], conv_b + s * 128, xc, Ks[s], Tks[s]);
    k_gnstats<<<dim3(8, B_), 256, 0, stream>>>(xc, stats, Tks[s]);
    int tot = B_ * H_ * N_ * Tks[s];
    k_gnact<<<(tot + 255) / 256, 256, 0, stream>>>(
        xc, stats, gn_pw + s * H_, gn_pb + s * H_, gn_qw + s * H_, gn_qb + s * H_,
        concat, Tks[s], offs[s], tot);
  }

  // fc + relu
  k_fc<<<(B_ * H_ * N_ + 255) / 256, 256, 0, stream>>>(concat, fc_w, fc_b, tco);

  // residual + relu + layernorm -> output [B,N,H,T]
  k_resln<<<dim3(N_, B_), 128, 0, stream>>>(x, tco, res_w, res_b, ln_w, ln_b,
                                            (float*)d_out);
}
